// SequentialMambaBlocks_84413287235936
// MI455X (gfx1250) — matrix operand order repaired and checked by reading, not compiled, not run on hardware
//
#include <hip/hip_runtime.h>
#include <hip/hip_bf16.h>
#include <cstdint>

#define DEPTH   12
#define D_MODEL 384
#define D_INNER 768
#define D_CONV  4
#define D_STATE 16
#define DT_RANK 24
#define NXP     (DT_RANK + 2 * D_STATE)   // 56
#define BB 2
#define TT 4
#define HP 14
#define WP 14
#define LL (TT * HP * WP)                 // 784
#define BL (BB * LL)                      // 1568

typedef __attribute__((ext_vector_type(16))) _Float16 v16h;
typedef __attribute__((ext_vector_type(8)))  _Float16 v8h;
typedef __attribute__((ext_vector_type(8)))  float    v8f;
typedef __attribute__((ext_vector_type(4)))  uint32_t v4u;

// ---------------------------------------------------------------------------
// f32 -> f16 conversion (weights)
// ---------------------------------------------------------------------------
__global__ void f32_to_f16_kernel(const float* __restrict__ src,
                                  _Float16* __restrict__ dst, int n) {
  int i = blockIdx.x * 256 + threadIdx.x;
  if (i < n) dst[i] = (_Float16)src[i];
}

// ---------------------------------------------------------------------------
// [B,T,C,H,W] -> hs[B*L, C]; res = 0
// ---------------------------------------------------------------------------
__global__ void reshape_in_kernel(const float* __restrict__ x,
                                  float* __restrict__ hs,
                                  float* __restrict__ res) {
  int i = blockIdx.x * 256 + threadIdx.x;
  if (i >= BL * D_MODEL) return;
  int c  = i % D_MODEL;
  int l  = (i / D_MODEL) % LL;
  int b  = i / (D_MODEL * LL);
  int t  = l / (HP * WP);
  int hw = l % (HP * WP);
  int hh = hw / WP, ww = hw % WP;
  hs[i]  = x[(((size_t)(b * TT + t) * D_MODEL + c) * HP + hh) * WP + ww];
  res[i] = 0.f;
}

// ---------------------------------------------------------------------------
// res += hs ; xn16 = f16(rmsnorm(res) * w)   (one block of 128 per row)
// ---------------------------------------------------------------------------
__global__ __launch_bounds__(128)
void addnorm_kernel(const float* __restrict__ hs, float* __restrict__ res,
                    const float* __restrict__ w, _Float16* __restrict__ xn16) {
  int row = blockIdx.x;
  int tid = threadIdx.x;
  __shared__ float red[128];
  const float* hr = hs + (size_t)row * D_MODEL;
  float* rr       = res + (size_t)row * D_MODEL;
  float v[3], ss = 0.f;
#pragma unroll
  for (int e = 0; e < 3; ++e) {
    int c = tid + e * 128;
    float t = rr[c] + hr[c];
    v[e] = t; ss += t * t;
  }
  red[tid] = ss;
  __syncthreads();
  for (int s = 64; s > 0; s >>= 1) {
    if (tid < s) red[tid] += red[tid + s];
    __syncthreads();
  }
  float scale = rsqrtf(red[0] / D_MODEL + 1e-5f);
#pragma unroll
  for (int e = 0; e < 3; ++e) {
    int c = tid + e * 128;
    rr[c] = v[e];
    xn16[(size_t)row * D_MODEL + c] = (_Float16)(v[e] * scale * w[c]);
  }
}

// ---------------------------------------------------------------------------
// WMMA GEMM:  C[M,N] = A[M,K] * W[N,K]^T  (A,W f16; C f32)
// Block: 256 thr = 8 waves; tile 32(M) x 128(N); K step 32.
// - Tiles staged with global_load_async_to_lds (ASYNCcnt)
// - B fragments read from W rows in LDS with plain b128 loads (no transpose)
// - each wave: 2x v_wmma_f32_16x16x32_f16 per K step (B-fragment reuse)
// ---------------------------------------------------------------------------
__global__ __launch_bounds__(256)
void gemm_f16_wmma_kernel(const _Float16* __restrict__ A,
                          const _Float16* __restrict__ W,
                          float* __restrict__ C, int M, int N, int K) {
  __shared__ _Float16 lA[32 * 32];
  __shared__ _Float16 lW[128 * 32];
  const int tid   = threadIdx.x;
  const int lane  = tid & 31;
  const int warp  = tid >> 5;
  const int m0    = blockIdx.y * 32;
  const int nblk  = blockIdx.x * 128;
  const int mrow  = lane & 15;
  const int khalf = (lane >> 4) << 3;   // 0 or 8 (A-frag K interleave)

  // staging geometry (fixed per thread)
  const int arow = tid >> 3;            // 0..31
  const int acol = (tid & 7) << 2;      // 0,4,..,28   (4 halves = b64)
  const int wrow = tid >> 1;            // 0..127
  const int wcol = (tid & 1) << 4;      // 0 or 16     (16 halves = 2x b128)
  const int wn   = nblk + wrow;
  const uint32_t la = (uint32_t)(uintptr_t)&lA[arow * 32 + acol];
  const uint32_t lw = (uint32_t)(uintptr_t)&lW[wrow * 32 + wcol];
  // transposed B-fragment source: 16x16 subtile of this wave's n-range
  const uint32_t btr =
      (uint32_t)(uintptr_t)&lW[(warp * 16 + (lane >> 1)) * 32] +
      ((lane & 1) << 4);

  // zero-fill pad rows once (async loads never touch n >= N rows)
  if (wn >= N) {
    uint4 z = {0, 0, 0, 0};
    *(uint4*)&lW[wrow * 32 + wcol]     = z;
    *(uint4*)&lW[wrow * 32 + wcol + 8] = z;
  }

  v8f acc0 = {}, acc1 = {};
  for (int k0 = 0; k0 < K; k0 += 32) {
    // async stage A tile 32x32 (b64 per thread, memory -> LDS, no VGPR trip)
    asm volatile("global_load_async_to_lds_b64 %0, %1, off"
                 :: "v"(la), "v"(&A[(size_t)(m0 + arow) * K + k0 + acol])
                 : "memory");
    // async stage W tile 128x32 (2x b128 per thread)
    if (wn < N) {
      const _Float16* gw = &W[(size_t)wn * K + k0 + wcol];
      asm volatile("global_load_async_to_lds_b128 %0, %1, off"
                   :: "v"(lw), "v"(gw) : "memory");
      asm volatile("global_load_async_to_lds_b128 %0, %1, off offset:16"
                   :: "v"(lw), "v"(gw) : "memory");
    }
    asm volatile("s_wait_asynccnt 0x0" ::: "memory");
    __syncthreads();

    // A fragments (two stacked 16-row tiles), contiguous b128 LDS reads
    v8h alo0 = *(const v8h*)&lA[mrow * 32 + khalf];
    v8h ahi0 = *(const v8h*)&lA[mrow * 32 + 16 + khalf];
    v8h alo1 = *(const v8h*)&lA[(16 + mrow) * 32 + khalf];
    v8h ahi1 = *(const v8h*)&lA[(16 + mrow) * 32 + 16 + khalf];
    v16h af0 = __builtin_shufflevector(alo0, ahi0,
        0, 1, 2, 3, 4, 5, 6, 7, 8, 9, 10, 11, 12, 13, 14, 15);
    v16h af1 = __builtin_shufflevector(alo1, ahi1,
        0, 1, 2, 3, 4, 5, 6, 7, 8, 9, 10, 11, 12, 13, 14, 15);

    // B fragment: row n = lane&15, K interleaved like A (khalf.., 16+khalf..)
    union { v4u u; v8h h; } c0, c1;
    c0.h = *(const v8h*)&lW[(warp * 16 + mrow) * 32 + khalf];
    c1.h = *(const v8h*)&lW[(warp * 16 + mrow) * 32 + 16 + khalf];
    asm volatile("s_wait_dscnt 0x0" ::: "memory");
    v16h bf = __builtin_shufflevector(c0.h, c1.h,
        0, 1, 2, 3, 4, 5, 6, 7, 8, 9, 10, 11, 12, 13, 14, 15);

    acc0 = __builtin_amdgcn_wmma_f32_16x16x32_f16(
        false, af0, false, bf, (short)0, acc0, false, false);
    acc1 = __builtin_amdgcn_wmma_f32_16x16x32_f16(
        false, af1, false, bf, (short)0, acc1, false, false);
    __syncthreads();
  }

  // C layout: lane&15 = N, (lane>>4)*8 + vgpr = M
  int cn = nblk + warp * 16 + (lane & 15);
  if (cn < N) {
    int mb = m0 + ((lane >> 4) << 3);
#pragma unroll
    for (int v = 0; v < 8; ++v)
      C[(size_t)(mb + v) * N + cn] = acc0[v];
#pragma unroll
    for (int v = 0; v < 8; ++v)
      C[(size_t)(mb + 16 + v) * N + cn] = acc1[v];
  }
}

// ---------------------------------------------------------------------------
// causal depthwise conv1d (k=4) + bias + SiLU over xx = xz[:, :768]
// ---------------------------------------------------------------------------
__global__ void conv_silu_kernel(const float* __restrict__ xz,
                                 const float* __restrict__ cw,
                                 const float* __restrict__ cb,
                                 float* __restrict__ xc,
                                 _Float16* __restrict__ xc16) {
  int i = blockIdx.x * 256 + threadIdx.x;
  if (i >= BL * D_INNER) return;
  int d  = i % D_INNER;
  int bt = i / D_INNER;
  int b  = bt / LL, l = bt % LL;
  float acc = cb[d];
#pragma unroll
  for (int k = 0; k < D_CONV; ++k) {
    int lt = l + k - (D_CONV - 1);
    if (lt >= 0)
      acc += cw[d * D_CONV + k] *
             xz[(size_t)(b * LL + lt) * (2 * D_INNER) + d];
  }
  float s = acc / (1.f + __expf(-acc));   // silu
  xc[i]   = s;
  xc16[i] = (_Float16)s;
}

// ---------------------------------------------------------------------------
// dt = softplus(dtl @ dtw^T + dtb)
// ---------------------------------------------------------------------------
__global__ void dt_kernel(const float* __restrict__ dbl,
                          const float* __restrict__ dtw,
                          const float* __restrict__ dtb,
                          float* __restrict__ dt) {
  int i = blockIdx.x * 256 + threadIdx.x;
  if (i >= BL * D_INNER) return;
  int d  = i % D_INNER;
  int bt = i / D_INNER;
  const float* r = dbl + (size_t)bt * NXP;
  float acc = dtb[d];
#pragma unroll
  for (int k = 0; k < DT_RANK; ++k) acc += r[k] * dtw[d * DT_RANK + k];
  dt[i] = (acc > 20.f) ? acc : log1pf(__expf(acc));
}

// ---------------------------------------------------------------------------
// selective scan: one thread per (b, d); 16 states in registers.
// y16 = f16( (sum_n h*C + xc*D) * silu(z) )
// grid = 6 blocks of 256 (b = blk/3, d = (blk%3)*256 + tid)
// ---------------------------------------------------------------------------
__global__ __launch_bounds__(256)
void scan_kernel(const float* __restrict__ dt, const float* __restrict__ xc,
                 const float* __restrict__ dbl, const float* __restrict__ xz,
                 const float* __restrict__ A_log, const float* __restrict__ Dp,
                 _Float16* __restrict__ y16) {
  int blk = blockIdx.x;
  int b   = blk / 3;
  int d   = (blk % 3) * 256 + threadIdx.x;
  __shared__ float sBC[NXP];
  float Ad[D_STATE];
#pragma unroll
  for (int n = 0; n < D_STATE; ++n) Ad[n] = -__expf(A_log[d * D_STATE + n]);
  float Dd = Dp[d];
  float h[D_STATE];
#pragma unroll
  for (int n = 0; n < D_STATE; ++n) h[n] = 0.f;

  for (int l = 0; l < LL; ++l) {
    int bt = b * LL + l;
    if (threadIdx.x < NXP) sBC[threadIdx.x] = dbl[(size_t)bt * NXP + threadIdx.x];
    __syncthreads();
    float dtv = dt[(size_t)bt * D_INNER + d];
    float xv  = xc[(size_t)bt * D_INNER + d];
    float zv  = xz[(size_t)bt * 2 * D_INNER + D_INNER + d];
    float y = 0.f;
#pragma unroll
    for (int n = 0; n < D_STATE; ++n) {
      float dA = __expf(dtv * Ad[n]);
      h[n] = h[n] * dA + dtv * sBC[DT_RANK + n] * xv;
      y += h[n] * sBC[DT_RANK + D_STATE + n];
    }
    y += xv * Dd;
    float sz = zv / (1.f + __expf(-zv));
    y16[(size_t)bt * D_INNER + d] = (_Float16)(y * sz);
    __syncthreads();
  }
}

// ---------------------------------------------------------------------------
// final rmsnorm(hs + res) * normf_w -> out[B,T,C,H,W]
// ---------------------------------------------------------------------------
__global__ __launch_bounds__(128)
void final_norm_kernel(const float* __restrict__ hs, const float* __restrict__ res,
                       const float* __restrict__ w, float* __restrict__ out) {
  int row = blockIdx.x;   // BL rows
  int tid = threadIdx.x;
  int b  = row / LL, l = row % LL;
  int t  = l / (HP * WP);
  int hw = l % (HP * WP);
  int hh = hw / WP, ww = hw % WP;
  __shared__ float red[128];
  float v[3], ss = 0.f;
#pragma unroll
  for (int e = 0; e < 3; ++e) {
    int c = tid + e * 128;
    float tv = hs[(size_t)row * D_MODEL + c] + res[(size_t)row * D_MODEL + c];
    v[e] = tv; ss += tv * tv;
  }
  red[tid] = ss;
  __syncthreads();
  for (int s = 64; s > 0; s >>= 1) {
    if (tid < s) red[tid] += red[tid + s];
    __syncthreads();
  }
  float scale = rsqrtf(red[0] / D_MODEL + 1e-5f);
#pragma unroll
  for (int e = 0; e < 3; ++e) {
    int c = tid + e * 128;
    out[(((size_t)(b * TT + t) * D_MODEL + c) * HP + hh) * WP + ww] =
        v[e] * scale * w[c];
  }
}

// ---------------------------------------------------------------------------
extern "C" void kernel_launch(void* const* d_in, const int* in_sizes, int n_in,
                              void* d_out, int out_size, void* d_ws, size_t ws_size,
                              hipStream_t stream) {
  const float* x        = (const float*)d_in[0];
  const float* norm_w   = (const float*)d_in[1];
  const float* in_proj  = (const float*)d_in[2];
  const float* conv_w   = (const float*)d_in[3];
  const float* conv_b   = (const float*)d_in[4];
  const float* x_proj   = (const float*)d_in[5];
  const float* dt_projw = (const float*)d_in[6];
  const float* dt_projb = (const float*)d_in[7];
  const float* A_log    = (const float*)d_in[8];
  const float* Dvec     = (const float*)d_in[9];
  const float* out_proj = (const float*)d_in[10];
  const float* normf_w  = (const float*)d_in[11];
  float* out = (float*)d_out;

  char* base = (char*)d_ws;
  size_t off = 0;
  auto alloc = [&](size_t bytes) -> void* {
    void* p = base + off;
    off = (off + bytes + 255) & ~(size_t)255;
    return p;
  };
  float*    hs    = (float*)   alloc((size_t)BL * D_MODEL * 4);
  float*    res   = (float*)   alloc((size_t)BL * D_MODEL * 4);
  _Float16* xn16  = (_Float16*)alloc((size_t)BL * D_MODEL * 2);
  float*    xz    = (float*)   alloc((size_t)BL * 2 * D_INNER * 4);
  float*    xc    = (float*)   alloc((size_t)BL * D_INNER * 4);
  _Float16* xc16  = (_Float16*)alloc((size_t)BL * D_INNER * 2);
  float*    dbl   = (float*)   alloc((size_t)BL * NXP * 4);
  float*    dtb_  = (float*)   alloc((size_t)BL * D_INNER * 4);
  _Float16* y16   = (_Float16*)alloc((size_t)BL * D_INNER * 2);
  _Float16* w16in = (_Float16*)alloc((size_t)2 * D_INNER * D_MODEL * 2);
  _Float16* w16xp = (_Float16*)alloc((size_t)NXP * D_INNER * 2);
  _Float16* w16op = (_Float16*)alloc((size_t)D_MODEL * D_INNER * 2);

  const int MT = BL / 32;   // 49 row tiles (32 M rows per block)

  reshape_in_kernel<<<(BL * D_MODEL + 255) / 256, 256, 0, stream>>>(x, hs, res);

  for (int i = 0; i < DEPTH; ++i) {
    int nIn = 2 * D_INNER * D_MODEL;   // 1536*384
    int nXp = NXP * D_INNER;           // 56*768
    int nOp = D_MODEL * D_INNER;       // 384*768
    f32_to_f16_kernel<<<(nIn + 255) / 256, 256, 0, stream>>>(
        in_proj + (size_t)i * nIn, w16in, nIn);
    f32_to_f16_kernel<<<(nXp + 255) / 256, 256, 0, stream>>>(
        x_proj + (size_t)i * nXp, w16xp, nXp);
    f32_to_f16_kernel<<<(nOp + 255) / 256, 256, 0, stream>>>(
        out_proj + (size_t)i * nOp, w16op, nOp);

    addnorm_kernel<<<BL, 128, 0, stream>>>(hs, res, norm_w + i * D_MODEL, xn16);

    // in_proj: [BL,384] x [1536,384]^T -> xz [BL,1536]
    gemm_f16_wmma_kernel<<<dim3(12, MT), 256, 0, stream>>>(
        xn16, w16in, xz, BL, 2 * D_INNER, D_MODEL);

    conv_silu_kernel<<<(BL * D_INNER + 255) / 256, 256, 0, stream>>>(
        xz, conv_w + (size_t)i * D_INNER * D_CONV, conv_b + (size_t)i * D_INNER,
        xc, xc16);

    // x_proj: [BL,768] x [56,768]^T -> dbl [BL,56]
    gemm_f16_wmma_kernel<<<dim3(1, MT), 256, 0, stream>>>(
        xc16, w16xp, dbl, BL, NXP, D_INNER);

    dt_kernel<<<(BL * D_INNER + 255) / 256, 256, 0, stream>>>(
        dbl, dt_projw + (size_t)i * D_INNER * DT_RANK,
        dt_projb + (size_t)i * D_INNER, dtb_);

    scan_kernel<<<6, 256, 0, stream>>>(
        dtb_, xc, dbl, xz, A_log + (size_t)i * D_INNER * D_STATE,
        Dvec + (size_t)i * D_INNER, y16);

    // out_proj: [BL,768] x [384,768]^T -> hs [BL,384]
    gemm_f16_wmma_kernel<<<dim3(3, MT), 256, 0, stream>>>(
        y16, w16op, hs, BL, D_MODEL, D_INNER);
  }

  final_norm_kernel<<<BL, 128, 0, stream>>>(hs, res, normf_w, out);
}
